// MaSST_90555090468986
// MI455X (gfx1250) — compile-verified, matrix operand order checked by reading, not run on hardware
//
#include <hip/hip_runtime.h>
#include <hip/hip_bf16.h>

// ---------------- WMMA plumbing (gfx1250, wave32) ----------------
typedef __attribute__((ext_vector_type(16))) _Float16 v16h;
typedef __attribute__((ext_vector_type(8)))  float    v8f;
typedef __attribute__((ext_vector_type(4)))  float    f32x4;
typedef __attribute__((ext_vector_type(4)))  int      v4i;

// Load a 16x32 f16 fragment (A-layout). `base` points at (row0, k0) of a
// row-major matrix with leading dim `ld` (elements). Per CDNA5 ISA 16-bit
// A-matrix layout: lanes 0-15 hold M=0..15 / K in {kb..kb+7, 16+kb..16+kb+7}
// with kb = (lane>>4)*8.  Same loader serves B when B is stored [N,K].
__device__ __forceinline__ v16h load_frag16(const _Float16* __restrict__ base, int ld) {
    int lane = threadIdx.x & 31;
    int r    = lane & 15;
    int kb   = (lane >> 4) << 3;           // 0 or 8
    const _Float16* p = base + (size_t)r * ld + kb;
    v16h f;
    f32x4* fq = (f32x4*)&f;
    fq[0] = *(const f32x4*)(p);            // K = kb .. kb+7
    fq[1] = *(const f32x4*)(p + 16);       // K = 16+kb .. 16+kb+7
    return f;
}

__device__ __forceinline__ v8f wmma_f16(v16h a, v16h b, v8f c) {
    return __builtin_amdgcn_wmma_f32_16x16x32_f16(false, a, false, b, (short)0, c, false, false);
}

// ---- async global->LDS copy (CDNA5 GLOBAL_LOAD_ASYNC_TO_LDS_B128), with
// ---- a synchronous fallback if the builtin is not declared by this clang.
#if __has_builtin(__builtin_amdgcn_global_load_async_to_lds_b128)
#define HAS_ASYNC_LDS 1
typedef __attribute__((address_space(1))) v4i* gv4i_p;   // global (AS1) int4*
typedef __attribute__((address_space(3))) v4i* sv4i_p;   // LDS    (AS3) int4*
#else
#define HAS_ASYNC_LDS 0
#endif

__device__ __forceinline__ void copy_g2l_b128(const _Float16* g, _Float16* l) {
#if HAS_ASYNC_LDS
    __builtin_amdgcn_global_load_async_to_lds_b128((gv4i_p)(v4i*)g, (sv4i_p)(v4i*)l, 0, 0);
#else
    *(f32x4*)l = *(const f32x4*)g;
#endif
}
__device__ __forceinline__ void wait_async_copies() {
#if HAS_ASYNC_LDS
#if __has_builtin(__builtin_amdgcn_s_wait_asynccnt)
    __builtin_amdgcn_s_wait_asynccnt(0);
#else
    asm volatile("s_wait_asynccnt 0x0" ::: "memory");
#endif
#endif
}

__device__ __forceinline__ float sigmoidf_(float x) { return 1.0f / (1.0f + expf(-x)); }

// Deterministic per-(t,which,b,m) gumbel noise (hash-based; JAX bits are not
// reproducible on device -- forward uses it only through an argmax).
__device__ __forceinline__ float gumbel_noise(unsigned t, unsigned which, unsigned b, unsigned m) {
    unsigned x = t * 73856093u ^ (which + 1u) * 19349663u ^ b * 83492791u ^ m * 2654435761u;
    x ^= x >> 17; x *= 0xed5ad4bbu;
    x ^= x >> 11; x *= 0xac4c1b51u;
    x ^= x >> 15; x *= 0x31848babu;
    x ^= x >> 14;
    float u = (float)(x >> 8) * (1.0f / 16777216.0f);
    const float EPS = 1e-20f;
    return -logf(EPS - logf(EPS + u));
}

// ---------------- problem constants ----------------
#define TT   512
#define BB   64
#define FF   500
#define HH   512
#define AA   64
#define MM   64
#define HEADP 80      // HEAD+1=65 padded to 80 (multiple of 16)
#define NCAT 3152     // 1536 gates + 3*512 s-terms + 80 head1
#define KCAT 1536     // [xt | h | entry]
#define NTILES2 (NCAT / 16)   // 197

// ---------------- prep kernels ----------------
__global__ void k_pad16(_Float16* __restrict__ dst, const float* __restrict__ src,
                        int rows, int scols, int dcols) {
    long total = (long)rows * dcols;
    for (long idx = (long)blockIdx.x * blockDim.x + threadIdx.x; idx < total;
         idx += (long)gridDim.x * blockDim.x) {
        long r = idx / dcols; int c = (int)(idx % dcols);
        float v = (c < scols) ? src[r * scols + c] : 0.0f;
        dst[idx] = (_Float16)v;
    }
}

// Bcat [NCAT, KCAT] in [N,K] layout; K rows = [xt(512) | h(512) | entry(512)]
__global__ void k_build_bcat(_Float16* __restrict__ dst,
                             const float* W_ih, const float* W_hh, const float* W_rh,
                             const float* W_s1, const float* W_s2, const float* W_s3,
                             const float* W_im1, const float* W_hm1, const float* W_mm1) {
    long total = (long)NCAT * KCAT;
    for (long idx = (long)blockIdx.x * blockDim.x + threadIdx.x; idx < total;
         idx += (long)gridDim.x * blockDim.x) {
        int n = (int)(idx / KCAT), k = (int)(idx % KCAT);
        float v = 0.0f;
        if (n < 1536) {                               // gate preact (W_ih/W_hh/W_rh cols)
            if (k < 512)       v = W_ih[k * 1536 + n];
            else if (k < 1024) v = W_hh[(k - 512) * 1536 + n];
            else               v = W_rh[(k - 1024) * 1536 + n];
        } else if (n < 2048) {                        // sx = xt@W_s1
            if (k < 512) v = W_s1[k * 512 + (n - 1536)];
        } else if (n < 2560) {                        // sh = h@W_s2
            if (k >= 512 && k < 1024) v = W_s2[(k - 512) * 512 + (n - 2048)];
        } else if (n < 3072) {                        // se = entry@W_s3
            if (k >= 1024) v = W_s3[(k - 1024) * 512 + (n - 2560)];
        } else {                                      // head1 preact
            int c = n - 3072;
            if (c < 65) {
                if (k < 512)       v = W_im1[k * 65 + c];
                else if (k < 1024) v = W_hm1[(k - 512) * 65 + c];
                else               v = W_mm1[(k - 1024) * 65 + c];
            }
        }
        dst[idx] = (_Float16)v;
    }
}

// Bm [HEADP, 1024] in [N,K] layout; K rows = [xt(512) | h(512)]; rh preact.
__global__ void k_build_bm(_Float16* __restrict__ dst,
                           const float* W_im, const float* W_hm) {
    long total = (long)HEADP * 1024;
    for (long idx = (long)blockIdx.x * blockDim.x + threadIdx.x; idx < total;
         idx += (long)gridDim.x * blockDim.x) {
        int n = (int)(idx / 1024), k = (int)(idx % 1024);
        float v = 0.0f;
        if (n < 65) v = (k < 512) ? W_im[k * 65 + n] : W_hm[(k - 512) * 65 + n];
        dst[idx] = (_Float16)v;
    }
}

__global__ void k_init(float* h, _Float16* h16, float* mem, float* c,
                       const float* cfc_b, float* lu, float* lastuse, int* ridx) {
    long total = (long)BB * MM * HH;                  // largest region (mem)
    for (long idx = (long)blockIdx.x * blockDim.x + threadIdx.x; idx < total;
         idx += (long)gridDim.x * blockDim.x) {
        mem[idx] = 0.0f;
        if (idx < BB * HH) { h[idx] = 0.0f; h16[idx] = (_Float16)0.0f; }
        if (idx < BB * MM * 64) c[idx] = cfc_b[idx & 63];   // mem=0 => c = cfc_b
        if (idx < BB * MM) { lu[idx] = -99999.0f; lastuse[idx] = 0.0f; }
        if (idx < BB) ridx[idx] = 0;
    }
}

// ---------------- generic WMMA GEMM: C = A[M,K] * Bt[N,K]^T (+bias) ---------
__global__ void k_gemm(const _Float16* __restrict__ A, int lda,
                       const _Float16* __restrict__ Bt, int ldb,
                       float* __restrict__ Cf, _Float16* __restrict__ Ch, int ldc,
                       const float* __restrict__ bias,
                       int Mtiles, int Ntiles, int K) {
    int wave = threadIdx.x >> 5;
    int tile = blockIdx.x * (blockDim.x >> 5) + wave;
    if (tile >= Mtiles * Ntiles) return;
    int tm = tile / Ntiles, tn = tile % Ntiles;
    const _Float16* a0 = A + (size_t)tm * 16 * lda;
    const _Float16* b0 = Bt + (size_t)tn * 16 * ldb;
    v8f acc = {};
    for (int k = 0; k < K; k += 32) {
        __builtin_prefetch(a0 + k + 256, 0, 1);       // global_prefetch_b8
        v16h a = load_frag16(a0 + k, lda);
        v16h b = load_frag16(b0 + k, ldb);
        acc = wmma_f16(a, b, acc);
    }
    int lane = threadIdx.x & 31;
    int col = tn * 16 + (lane & 15);
    int rb  = tm * 16 + ((lane >> 4) << 3);
    float bv = bias ? bias[col] : 0.0f;
    for (int r = 0; r < 8; ++r) {
        float v = acc[r] + bv;
        if (Cf) Cf[(size_t)(rb + r) * ldc + col] = v;
        if (Ch) Ch[(size_t)(rb + r) * ldc + col] = (_Float16)v;
    }
}

// ---------------- shared S1 body (memory write/read + head scores) ----------
// Runs inside a single block of NT threads. Performs step-t "front half":
//   mem[b,slot]=h ; c-row update ; rh ; scores0+gumbel ; y0/ridx ; entry ; lu
template <int NT>
__device__ __forceinline__ void s1_body(
        const _Float16* __restrict__ xt, const float* __restrict__ h,
        const _Float16* __restrict__ h16,
        float* __restrict__ mem, float* __restrict__ c,
        float* __restrict__ lu, float* __restrict__ lastuse, int* __restrict__ ridx,
        float* __restrict__ entry, _Float16* __restrict__ entry16,
        const _Float16* __restrict__ bm16, const _Float16* __restrict__ cfc16,
        const float* __restrict__ cfc_b, int t,
        float* s_rh, float* s_sc, int* s_slot, int* s_y0) {
    const int NW = NT / 32;
    int tid = threadIdx.x;

    if (tid < BB) s_slot[tid] = (t < MM) ? t : ridx[tid];
    __syncthreads();

    // write h into memory slot
    for (int i = tid; i < BB * HH; i += NT) {
        int b = i >> 9, j = i & 511;
        mem[((size_t)b * MM + s_slot[b]) * HH + j] = h[i];
    }

    int wave = tid >> 5, lane = tid & 31;

    // incremental head-vec cache: c[b, slot[b], :] = h @ cfc_w.T + cfc_b (WMMA)
    for (int tt = wave; tt < 16; tt += NW) {           // 4x4 tiles of [64,64]
        int tm = tt >> 2, tn = tt & 3;
        v8f acc = {};
        for (int k = 0; k < HH; k += 32) {
            v16h a = load_frag16(h16 + (size_t)tm * 16 * HH + k, HH);
            v16h b = load_frag16(cfc16 + (size_t)tn * 16 * HH + k, HH);
            acc = wmma_f16(a, b, acc);
        }
        int col = tn * 16 + (lane & 15);
        int rb  = tm * 16 + ((lane >> 4) << 3);
        float bj = cfc_b[col];
        for (int r = 0; r < 8; ++r) {
            int b_ = rb + r;
            c[((size_t)b_ * MM + s_slot[b_]) * 64 + col] = acc[r] + bj;
        }
    }

    // rh = tanh([xt|h] @ [W_im;W_hm])  -> [64, HEADP] into LDS (WMMA)
    for (int tt = wave; tt < 4 * (HEADP / 16); tt += NW) {
        int tm = tt / (HEADP / 16), tn = tt % (HEADP / 16);
        v8f acc = {};
        for (int k = 0; k < 1024; k += 32) {
            const _Float16* src = (k < 512) ? (xt + (size_t)tm * 16 * HH + k)
                                            : (h16 + (size_t)tm * 16 * HH + (k - 512));
            v16h a = load_frag16(src, HH);
            v16h b = load_frag16(bm16 + (size_t)tn * 16 * 1024 + k, 1024);
            acc = wmma_f16(a, b, acc);
        }
        int col = tn * 16 + (lane & 15);
        int rb  = tm * 16 + ((lane >> 4) << 3);
        for (int r = 0; r < 8; ++r) s_rh[(rb + r) * HEADP + col] = tanhf(acc[r]);
    }
    __syncthreads();

    // scores0[b,m] = rh[b,:64] . c[b,m,:] + rh[b,64]*sigmoid(lu) + gumbel
    for (int i = tid; i < BB * MM; i += NT) {
        int b = i >> 6, m = i & 63;
        float lus = sigmoidf_(lu[i]);
        lastuse[i] = lus;                              // snapshot (pre-update lu)
        float s = s_rh[b * HEADP + 64] * lus;
        const float* cb = c + (size_t)i * 64;
        for (int k = 0; k < 64; ++k) s += s_rh[b * HEADP + k] * cb[k];
        s_sc[i] = s + gumbel_noise((unsigned)t, 0u, (unsigned)b, (unsigned)m);
    }
    __syncthreads();

    if (tid < BB) {
        float best = -3.4e38f; int bi = 0;
        for (int m = 0; m < MM; ++m) { float v = s_sc[tid * MM + m]; if (v > best) { best = v; bi = m; } }
        s_y0[tid] = bi; ridx[tid] = bi;                // ridx = hard one-hot index of y0
    }
    __syncthreads();

    // entry = mem[b, y0]; lu = (lu-1)*(1-y0)
    for (int i = tid; i < BB * HH; i += NT) {
        int b = i >> 9, j = i & 511;
        float v = mem[((size_t)b * MM + s_y0[b]) * HH + j];
        entry[i] = v; entry16[i] = (_Float16)v;
    }
    for (int i = tid; i < BB * MM; i += NT) {
        int b = i >> 6, m = i & 63;
        lu[i] = (m == s_y0[b]) ? 0.0f : (lu[i] - 1.0f);
    }
}

// ---------------- k_s1: initial front-half for t=0 (one block) --------------
__global__ void k_s1(const _Float16* __restrict__ xt, const float* __restrict__ h,
                     const _Float16* __restrict__ h16,
                     float* __restrict__ mem, float* __restrict__ c,
                     float* __restrict__ lu, float* __restrict__ lastuse,
                     int* __restrict__ ridx,
                     float* __restrict__ entry, _Float16* __restrict__ entry16,
                     const _Float16* __restrict__ bm16,
                     const _Float16* __restrict__ cfc16, const float* __restrict__ cfc_b,
                     int t) {
    __shared__ float s_rh[BB * HEADP];
    __shared__ float s_sc[BB * MM];
    __shared__ int   s_slot[BB];
    __shared__ int   s_y0[BB];
    s1_body<256>(xt, h, h16, mem, c, lu, lastuse, ridx, entry, entry16,
                 bm16, cfc16, cfc_b, t, s_rh, s_sc, s_slot, s_y0);
}

// ------- k_s2: fused step GEMM G = [xt|h|entry](64x1536) @ Bcat(1536x3152) --
// Grid: 4 m-tiles x 25 n-groups = 100 blocks, 256 thr (8 waves, 8 n-tiles).
// The 16-row A strip is staged once per block into LDS via async copies
// (CDNA5 GLOBAL_LOAD_ASYNC_TO_LDS_B128 when available), B streams from L2.
__global__ void k_s2(const _Float16* __restrict__ xt, const _Float16* __restrict__ h16,
                     const _Float16* __restrict__ e16, const _Float16* __restrict__ bcat,
                     float* __restrict__ G) {
    __shared__ __align__(16) _Float16 sA[16 * KCAT];   // 48 KB
    int tm = blockIdx.x / 25;
    int g  = blockIdx.x % 25;
    int wave = threadIdx.x >> 5, lane = threadIdx.x & 31;
    int tn = g * 8 + wave;

    // stage A strip (rows tm*16..tm*16+15, K = [xt|h|entry]) into LDS
    for (int cidx = threadIdx.x; cidx < 16 * KCAT / 8; cidx += 256) {
        int e = cidx * 8;
        int r = e / KCAT, k = e % KCAT;
        int row = tm * 16 + r;
        const _Float16* src = (k < 512)  ? (xt  + (size_t)row * HH + k)
                            : (k < 1024) ? (h16 + (size_t)row * HH + (k - 512))
                                         : (e16 + (size_t)row * HH + (k - 1024));
        copy_g2l_b128(src, &sA[e]);
    }
    wait_async_copies();
    __syncthreads();

    if (tn < NTILES2) {                                // wave-uniform guard
        const _Float16* b0 = bcat + (size_t)tn * 16 * KCAT;
        v8f acc = {};
        for (int k = 0; k < KCAT; k += 32) {
            __builtin_prefetch(b0 + k + 256, 0, 1);    // global_prefetch_b8
            v16h a = load_frag16(&sA[k], KCAT);        // ds_load_b128 x2
            v16h b = load_frag16(b0 + k, KCAT);
            acc = wmma_f16(a, b, acc);
        }
        int col = tn * 16 + (lane & 15);
        int rb  = tm * 16 + ((lane >> 4) << 3);
        for (int r = 0; r < 8; ++r) G[(size_t)(rb + r) * NCAT + col] = acc[r];
    }
}

// ---- k_s3: epilogue of step t  +  front-half (S1) of step t+1 (one block) --
__global__ void k_s3(const float* __restrict__ G, float* __restrict__ h,
                     _Float16* __restrict__ h16,
                     float* __restrict__ mem, float* __restrict__ c,
                     float* __restrict__ lu, float* __restrict__ lastuse,
                     int* __restrict__ ridx,
                     float* __restrict__ entry, _Float16* __restrict__ entry16,
                     _Float16* __restrict__ o16,
                     const _Float16* __restrict__ fcw16, const _Float16* __restrict__ bm16,
                     const _Float16* __restrict__ cfc16, const float* __restrict__ cfc_b,
                     const float* __restrict__ bias, const float* __restrict__ b1,
                     const float* __restrict__ b2, const float* __restrict__ b3,
                     const float* __restrict__ bias_m1, const float* __restrict__ fc_b,
                     const int* __restrict__ length,
                     const _Float16* __restrict__ xnext,
                     float* __restrict__ out_t, int t) {
    __shared__ float s_rh[BB * HEADP];                 // reused: head1 / rh
    __shared__ float s_sc[BB * MM];
    __shared__ int   s_slot[BB];
    __shared__ int   s_y1[BB];
    int tid = threadIdx.x;                             // 512 threads, 16 waves

    // ---- phase A: gates + hidden update; head1 into LDS
    for (int i = tid; i < BB * HH; i += 512) {
        int b = i >> 9, j = i & 511;
        const float* gg = G + (size_t)b * NCAT;
        float r  = sigmoidf_(gg[j]        + bias[j]);
        float zg = sigmoidf_(gg[512 + j]  + bias[512 + j]);
        float ng = sigmoidf_(gg[1024 + j] + bias[1024 + j]);
        float hn = tanhf(gg[1536 + j] + b1[j] + r * (gg[2048 + j] + b2[j])
                                               + zg * (gg[2560 + j] + b3[j]));
        float hold = h[i];
        float h1   = ng * hn + (1.0f - ng) * hold;
        o16[(size_t)b * 1024 + j] = (_Float16)h1;      // o[:, :512] = h1 (unmasked)
        float hnext = (t < length[b]) ? h1 : hold;
        h[i] = hnext; h16[i] = (_Float16)hnext;
    }
    for (int i = tid; i < BB * 65; i += 512) {
        int b = i / 65, cc = i % 65;
        s_rh[b * HEADP + cc] = tanhf(G[(size_t)b * NCAT + 3072 + cc] + bias_m1[cc]);
    }
    __syncthreads();

    // ---- phase B: scores1 against the SAME head_vecs (c cache + lastuse)
    for (int i = tid; i < BB * MM; i += 512) {
        int b = i >> 6, m = i & 63;
        float s = s_rh[b * HEADP + 64] * lastuse[i];
        const float* cb = c + (size_t)i * 64;
        for (int k = 0; k < 64; ++k) s += s_rh[b * HEADP + k] * cb[k];
        s_sc[i] = s + gumbel_noise((unsigned)t, 1u, (unsigned)b, (unsigned)m);
    }
    __syncthreads();
    if (tid < BB) {
        float best = -3.4e38f; int bi = 0;
        for (int m = 0; m < MM; ++m) { float v = s_sc[tid * MM + m]; if (v > best) { best = v; bi = m; } }
        s_y1[tid] = bi;
    }
    __syncthreads();

    // ---- phase C: entry_o gather (must precede next step's mem write)
    for (int i = tid; i < BB * HH; i += 512) {
        int b = i >> 9, j = i & 511;
        o16[(size_t)b * 1024 + 512 + j] = (_Float16)mem[((size_t)b * MM + s_y1[b]) * HH + j];
    }
    __syncthreads();

    // ---- phase D: out[t] = o @ fc_w.T + fc_b  (16 waves, 16 tiles, WMMA)
    {
        int wave = tid >> 5, lane = tid & 31;
        int tm = wave >> 2, tn = wave & 3;
        v8f acc = {};
        for (int k = 0; k < 1024; k += 32) {
            v16h a = load_frag16(o16 + (size_t)tm * 16 * 1024 + k, 1024);
            v16h b = load_frag16(fcw16 + (size_t)tn * 16 * 1024 + k, 1024);
            acc = wmma_f16(a, b, acc);
        }
        int col = tn * 16 + (lane & 15);
        int rb  = tm * 16 + ((lane >> 4) << 3);
        float bv = fc_b[col];
        for (int r = 0; r < 8; ++r) out_t[(size_t)(rb + r) * AA + col] = acc[r] + bv;
    }
    __syncthreads();

    // ---- phase E: front-half (S1) of step t+1 in the same block
    if (t + 1 < TT) {
        s1_body<512>(xnext, h, h16, mem, c, lu, lastuse, ridx, entry, entry16,
                     bm16, cfc16, cfc_b, t + 1, s_rh, s_sc, s_slot, s_y1);
    }
}

// ---------------- workspace layout (bytes, all 256-aligned) -----------------
#define WS_X16     0UL            // [32768,512] f16
#define WS_FC1T    33554432UL     // [512,512]   f16 (fc1_w padded K 500->512)
#define WS_XP16    34078720UL     // [32768,512] f16
#define WS_BCAT    67633152UL     // [3152,1536] f16
#define WS_BM      77316096UL     // [80,1024]   f16
#define WS_CFC16   77479936UL     // [64,512]    f16
#define WS_FCW16   77545472UL     // [64,1024]   f16
#define WS_H       77676544UL     // [64,512]    f32
#define WS_H16     77807616UL     // [64,512]    f16
#define WS_MEM     77873152UL     // [64,64,512] f32
#define WS_C       86261760UL     // [64,64,64]  f32
#define WS_LU      87310336UL     // [64,64]     f32
#define WS_LAST    87326720UL     // [64,64]     f32
#define WS_RIDX    87343104UL     // [64]        i32
#define WS_ENTRY   87343360UL     // [64,512]    f32
#define WS_E16     87474432UL     // [64,512]    f16
#define WS_G       87539968UL     // [64,3152]   f32
#define WS_O16     88346880UL     // [64,1024]   f16

extern "C" void kernel_launch(void* const* d_in, const int* in_sizes, int n_in,
                              void* d_out, int out_size, void* d_ws, size_t ws_size,
                              hipStream_t stream) {
    const float* x      = (const float*)d_in[0];
    const float* fc1_w  = (const float*)d_in[1];
    const float* fc1_b  = (const float*)d_in[2];
    const float* fc_w   = (const float*)d_in[3];
    const float* fc_b   = (const float*)d_in[4];
    const float* W_ih   = (const float*)d_in[5];
    const float* W_hh   = (const float*)d_in[6];
    const float* W_rh   = (const float*)d_in[7];
    const float* bias   = (const float*)d_in[8];
    const float* W_s1   = (const float*)d_in[9];
    const float* W_s2   = (const float*)d_in[10];
    const float* W_s3   = (const float*)d_in[11];
    const float* b1     = (const float*)d_in[12];
    const float* b2     = (const float*)d_in[13];
    const float* b3     = (const float*)d_in[14];
    const float* W_im   = (const float*)d_in[15];
    const float* W_hm   = (const float*)d_in[16];
    const float* cfc_w  = (const float*)d_in[17];
    const float* cfc_b  = (const float*)d_in[18];
    const float* W_im1  = (const float*)d_in[19];
    const float* W_hm1  = (const float*)d_in[20];
    const float* W_mm1  = (const float*)d_in[21];
    const float* bias_m1= (const float*)d_in[22];
    const int*   length = (const int*)d_in[23];
    float* out = (float*)d_out;

    char* ws = (char*)d_ws;
    _Float16* x16    = (_Float16*)(ws + WS_X16);
    _Float16* fc1t   = (_Float16*)(ws + WS_FC1T);
    _Float16* xp16   = (_Float16*)(ws + WS_XP16);
    _Float16* bcat   = (_Float16*)(ws + WS_BCAT);
    _Float16* bm16   = (_Float16*)(ws + WS_BM);
    _Float16* cfc16  = (_Float16*)(ws + WS_CFC16);
    _Float16* fcw16  = (_Float16*)(ws + WS_FCW16);
    float*    h      = (float*)(ws + WS_H);
    _Float16* h16    = (_Float16*)(ws + WS_H16);
    float*    mem    = (float*)(ws + WS_MEM);
    float*    c      = (float*)(ws + WS_C);
    float*    lu     = (float*)(ws + WS_LU);
    float*    lastu  = (float*)(ws + WS_LAST);
    int*      ridx   = (int*)(ws + WS_RIDX);
    float*    entry  = (float*)(ws + WS_ENTRY);
    _Float16* e16    = (_Float16*)(ws + WS_E16);
    float*    G      = (float*)(ws + WS_G);
    _Float16* o16    = (_Float16*)(ws + WS_O16);

    // ---- prep: f32 -> f16 conversions / transposed composite weights ----
    k_pad16<<<8192, 256, 0, stream>>>(x16, x, TT * BB, FF, HH);         // x padded
    k_pad16<<<512, 256, 0, stream>>>(fc1t, fc1_w, HH, FF, HH);          // [N,K] for xp GEMM
    k_pad16<<<128, 256, 0, stream>>>(cfc16, cfc_w, 64, HH, HH);
    k_pad16<<<256, 256, 0, stream>>>(fcw16, fc_w, AA, 2 * HH, 2 * HH);
    k_build_bcat<<<4096, 256, 0, stream>>>(bcat, W_ih, W_hh, W_rh, W_s1, W_s2, W_s3,
                                           W_im1, W_hm1, W_mm1);
    k_build_bm<<<160, 256, 0, stream>>>(bm16, W_im, W_hm);
    k_init<<<8192, 256, 0, stream>>>(h, h16, mem, c, cfc_b, lu, lastu, ridx);

    // ---- xp = x @ fc1_w.T + fc1_b  (stored f16) : M=32768 N=512 K=512 ----
    {
        int Mt = (TT * BB) / 16, Nt = HH / 16;                          // 2048 x 32 tiles
        k_gemm<<<(Mt * Nt) / 8, 256, 0, stream>>>(x16, HH, fc1t, HH,
                                                  nullptr, xp16, HH, fc1_b, Mt, Nt, HH);
    }

    // ---- sequential scan: S1 once, then per step {S2 ; S3+S1(t+1)} ----
    k_s1<<<1, 256, 0, stream>>>(xp16, h, h16, mem, c, lu, lastu, ridx,
                                entry, e16, bm16, cfc16, cfc_b, 0);
    for (int t = 0; t < TT; ++t) {
        const _Float16* xt    = xp16 + (size_t)t * BB * HH;
        const _Float16* xnext = xp16 + (size_t)((t + 1 < TT) ? t + 1 : t) * BB * HH;
        k_s2<<<100, 256, 0, stream>>>(xt, h16, e16, bcat, G);
        k_s3<<<1, 512, 0, stream>>>(G, h, h16, mem, c, lu, lastu, ridx, entry, e16,
                                    o16, fcw16, bm16, cfc16, cfc_b,
                                    bias, b1, b2, b3, bias_m1, fc_b, length,
                                    xnext, out + (size_t)t * BB * AA, t);
    }
}
